// FalconAttention_86225763434784
// MI455X (gfx1250) — matrix operand order repaired and checked by reading, not compiled, not run on hardware
//
#include <hip/hip_runtime.h>
#include <hip/hip_bf16.h>
#include <cstdint>
#include <cstddef>

// ---------------- problem dims (compile-time) ----------------
#define BB   4
#define SS   1024
#define HH   32
#define HKVN 8
#define DD   128
#define GG   4          // H / HKV
#define HIDN 4096       // H*D
#define NQKV 6144       // (H + 2*HKV)*D
#define TT   4096       // B*S

// ---------------- TDM availability ----------------
#if defined(__has_builtin)
# if __has_builtin(__builtin_amdgcn_tensor_load_to_lds) && __has_builtin(__builtin_amdgcn_s_wait_tensorcnt)
#  define USE_TDM 1
# endif
#endif
#ifndef USE_TDM
# define USE_TDM 0
#endif

// ---------------- vector / fragment types ----------------
typedef __attribute__((ext_vector_type(16))) __bf16          v16bf;
typedef __attribute__((ext_vector_type(8)))  float           v8f;
typedef __attribute__((ext_vector_type(8)))  unsigned short  u16x8;
typedef __attribute__((ext_vector_type(4)))  unsigned short  u16x4;
typedef __attribute__((ext_vector_type(4)))  unsigned int    u32x4;
typedef __attribute__((ext_vector_type(8)))  int             i32x8;
typedef __attribute__((ext_vector_type(4)))  int             i32x4;

union Frag16 {
    v16bf v;
    u16x8 h[2];
    unsigned short e[16];
};

__device__ __forceinline__ unsigned short f2bf(float f) {
    unsigned int u = __float_as_uint(f);
    u += 0x7fffu + ((u >> 16) & 1u);          // round-to-nearest-even
    return (unsigned short)(u >> 16);
}

__device__ __forceinline__ v8f wmma_bf16(const v16bf& a, const v16bf& b, const v8f& c) {
    return __builtin_amdgcn_wmma_f32_16x16x32_bf16(false, a, false, b, (short)0, c, false, false);
}

// ---------------- kernel 1: f32 -> bf16 convert ----------------
__global__ __launch_bounds__(256) void cvt_f32_bf16_kernel(
    const float* __restrict__ in, unsigned short* __restrict__ out, int n)
{
    int i = (blockIdx.x * 256 + threadIdx.x) * 4;
    if (i + 3 < n) {
        float4 f = *(const float4*)(in + i);
        u16x4 o;
        o.x = f2bf(f.x); o.y = f2bf(f.y); o.z = f2bf(f.z); o.w = f2bf(f.w);
        *(u16x4*)(out + i) = o;
    }
}

// ---------------- kernel 2: bf16 WMMA GEMM, C[M,N] = A[M,K] @ W[N,K]^T ----------------
// Block: 128 threads = 4 waves; block tile 128x128; each wave computes 64x64
// (4x4 grid of 16x16 WMMA accumulators). Fragment layouts follow
// CDNA5 ISA 7.12.2 (16-bit A 16x32, B 32x16, f32 C/D 16x16).
__global__ __launch_bounds__(128) void gemm_bf16_nt_kernel(
    const unsigned short* __restrict__ A,   // M x K, row-major
    const unsigned short* __restrict__ W,   // N x K, row-major
    float* __restrict__ C,                  // M x N
    int M, int N, int K)
{
    const int lane = threadIdx.x & 31;
    const int wave = threadIdx.x >> 5;          // 0..3
    const int half = lane >> 4;                 // 0/1
    const int l16  = lane & 15;

    const int m0 = blockIdx.y * 128 + (wave >> 1) * 64;
    const int n0 = blockIdx.x * 128 + (wave & 1) * 64;

    v8f acc[4][4];
#pragma unroll
    for (int i = 0; i < 4; ++i)
#pragma unroll
        for (int j = 0; j < 4; ++j)
#pragma unroll
            for (int e = 0; e < 8; ++e) acc[i][j][e] = 0.0f;

    for (int kb = 0; kb < K; kb += 32) {
        Frag16 af[4], bf[4];
#pragma unroll
        for (int i = 0; i < 4; ++i) {
            // A frag: lane holds row m, K = kb + half*8 + {0..7} and +16
            const unsigned short* ap = A + (size_t)(m0 + i * 16 + l16) * K + kb + half * 8;
            af[i].h[0] = *(const u16x8*)(ap);
            af[i].h[1] = *(const u16x8*)(ap + 16);
        }
#pragma unroll
        for (int j = 0; j < 4; ++j) {
            // B frag: lane holds col n, K = kb + half*8 + {0..7} and +16
            const unsigned short* wp = W + (size_t)(n0 + j * 16 + l16) * K + kb + half * 8;
            bf[j].h[0] = *(const u16x8*)(wp);
            bf[j].h[1] = *(const u16x8*)(wp + 16);
        }
#pragma unroll
        for (int i = 0; i < 4; ++i)
#pragma unroll
            for (int j = 0; j < 4; ++j)
                acc[i][j] = wmma_bf16(af[i].v, bf[j].v, acc[i][j]);
    }

    // C/D layout: lane col n = l16; VGPR r -> row r + 8*half
#pragma unroll
    for (int i = 0; i < 4; ++i)
#pragma unroll
        for (int j = 0; j < 4; ++j) {
            const int col = n0 + j * 16 + l16;
#pragma unroll
            for (int r = 0; r < 8; ++r) {
                const int row = m0 + i * 16 + half * 8 + r;
                C[(size_t)row * N + col] = acc[i][j][r];
            }
        }
}

// ---------------- kernel 3: RoPE + split + bf16 cast ----------------
__global__ __launch_bounds__(256) void rope_split_kernel(
    const float* __restrict__ qkv,          // [T][6144] f32
    const int* __restrict__ positions,      // [T]
    unsigned short* __restrict__ Qo,        // [T][H][D]   bf16 (roped)
    unsigned short* __restrict__ Ko,        // [T][HKV][D] bf16 (roped)
    unsigned short* __restrict__ Vo)        // [T][HKV][D] bf16
{
    const int t = blockIdx.x;
    const float p = (float)positions[t];
    const float* base = qkv + (size_t)t * NQKV;
    const float kfreq = -0.14391156831212787f;   // -ln(10000)/64

    for (int i = threadIdx.x; i < HH * 64; i += 256) {
        const int head = i >> 6, d = i & 63;
        const float fr = p * __expf(kfreq * (float)d);
        const float cf = __cosf(fr), sf = __sinf(fr);
        const float x1 = base[head * DD + d];
        const float x2 = base[head * DD + 64 + d];
        unsigned short* q = Qo + (size_t)t * (HH * DD) + head * DD;
        q[d]      = f2bf(x1 * cf - x2 * sf);
        q[64 + d] = f2bf(x2 * cf + x1 * sf);
    }
    const float* kin = base + HH * DD;
    for (int i = threadIdx.x; i < HKVN * 64; i += 256) {
        const int head = i >> 6, d = i & 63;
        const float fr = p * __expf(kfreq * (float)d);
        const float cf = __cosf(fr), sf = __sinf(fr);
        const float x1 = kin[head * DD + d];
        const float x2 = kin[head * DD + 64 + d];
        unsigned short* k = Ko + (size_t)t * (HKVN * DD) + head * DD;
        k[d]      = f2bf(x1 * cf - x2 * sf);
        k[64 + d] = f2bf(x2 * cf + x1 * sf);
    }
    const float* vin = base + (HH + HKVN) * DD;
    for (int i = threadIdx.x; i < HKVN * DD; i += 256)
        Vo[(size_t)t * (HKVN * DD) + i] = f2bf(vin[i]);
}

// ---------------- kernel 4: causal GQA attention ----------------
// Grid: (S*G/64, HKV, B). Block: 128 threads (4 waves).
// LDS plan (CDNA5: 320 KB/WGP):
//   scores  64 x 1028 f32                = 263168 B  (one-pass softmax, no rescale)
//   vt      128 x 72 u16 (V^T staging)   =  18432 B
//   kstage  64 x 136 u16 (TDM K tile)    =  17408 B  (272B row stride via TDM pad)
#define SC_STRIDE 1028
#define VT_STRIDE 72
#define KSTRIDE   136            // 128 elems + 8 pad elems (16B) per key row
#define SC_BYTES  (64 * SC_STRIDE * 4)
#define VT_BYTES  (128 * VT_STRIDE * 2)
#define KS_BYTES  (64 * KSTRIDE * 2)
#define ATTN_LDS_BYTES (SC_BYTES + VT_BYTES + KS_BYTES)

__global__ __launch_bounds__(128) void falcon_attention_kernel(
    const unsigned short* __restrict__ Q,    // [T][H][D]
    const unsigned short* __restrict__ Kb,   // [T][HKV][D]
    const unsigned short* __restrict__ Vb,   // [T][HKV][D]
    unsigned short* __restrict__ O)          // [T][HID]
{
    extern __shared__ char smem[];
    float* sc = (float*)smem;
    unsigned short* vt     = (unsigned short*)(smem + SC_BYTES);
    unsigned short* kstage = (unsigned short*)(smem + SC_BYTES + VT_BYTES);

    const int tile = blockIdx.x;
    const int h    = blockIdx.y;
    const int b    = blockIdx.z;
    const int tid  = threadIdx.x;
    const int lane = tid & 31, wave = tid >> 5;
    const int half = lane >> 4, l16 = lane & 15;

    const int r0 = tile * 64;               // Q_eff rows; s in [tile*16, tile*16+16)
    const int nk = tile * 16 + 16;          // max valid keys
    const int nk64 = ((nk + 63) >> 6) * 64;
    const float scale = 0.08838834764831845f;  // 1/sqrt(128)

    // ---- preload Q fragments for this wave's 16 rows (live in VGPRs throughout) ----
    Frag16 qa[4];
    {
        const int r = r0 + wave * 16 + l16;
        const int s = r >> 2, g = r & 3;
        const unsigned short* qp = Q + ((size_t)(b * SS + s) * HH + h * GG + g) * DD;
#pragma unroll
        for (int c = 0; c < 4; ++c) {
            qa[c].h[0] = *(const u16x8*)(qp + c * 32 + half * 8);
            qa[c].h[1] = *(const u16x8*)(qp + c * 32 + half * 8 + 16);
        }
    }

    // ---- scores = scale * Q @ K^T into LDS; K tile staged by the Tensor Data Mover ----
    for (int t0 = 0; t0 < nk64; t0 += 64) {
#if USE_TDM
        if (wave == 0) {
            // D# per CDNA5 ISA: 2D tile, 64 keys x 128 bf16,
            // row stride HKV*D elems; LDS pad 16B per 256B row -> 272B LDS row stride.
            const unsigned long long ga =
                (unsigned long long)(uintptr_t)(Kb + ((size_t)(b * SS + t0) * HKVN + h) * DD);
            const unsigned int lds_off = (unsigned int)(uintptr_t)kstage; // LDS aperture: low 32 bits
            u32x4 g0;
            g0[0] = 1u;                                        // count=1 (valid user D#)
            g0[1] = lds_off;                                   // lds_addr
            g0[2] = (unsigned int)ga;                          // global_addr[31:0]
            g0[3] = (unsigned int)((ga >> 32) & 0x1FFFFFFu)    // global_addr[56:32]
                  | (2u << 30);                                // type = 2 ("image")
            i32x8 g1;
            g1[0] = (1 << 16)        // data_size = 1 -> 2 bytes
                  | (1 << 20)        // pad_enable
                  | (5 << 22)        // pad_interval: 64 DWORDs (256B) between pads
                  | (3 << 25);       // pad_amount: 4 DWORDs (16B)
            g1[1] = (DD & 0xffff) << 16;          // tensor_dim0 = 128
            g1[2] = (SS & 0xffff) << 16;          // tensor_dim1 = 1024 (low 16)
            g1[3] = (DD & 0xffff) << 16;          // tile_dim0 = 128
            g1[4] = 64;                           // tile_dim1 = 64 keys, tile_dim2 = 0
            g1[5] = HKVN * DD;                    // tensor_dim0_stride = 1024 elems
            g1[6] = 0;
            g1[7] = 0;
            i32x4 gz = {0, 0, 0, 0};
#if __clang_major__ >= 23
            i32x8 gz8 = {0, 0, 0, 0, 0, 0, 0, 0};
            __builtin_amdgcn_tensor_load_to_lds(g0, g1, gz, gz, gz8, 0);
#else
            __builtin_amdgcn_tensor_load_to_lds(g0, g1, gz, gz, 0);
#endif
            __builtin_amdgcn_s_wait_tensorcnt(0);
        }
#else
        // fallback: cooperative b128 copy of the K tile into the padded LDS layout
        for (int c = tid; c < 64 * 16; c += 128) {
            const int kr = c >> 4, ch = c & 15;
            const unsigned short* src =
                Kb + ((size_t)(b * SS + t0 + kr) * HKVN + h) * DD + ch * 8;
            *(u16x8*)(kstage + kr * KSTRIDE + ch * 8) = *(const u16x8*)src;
        }
#endif
        __syncthreads();

        v8f acc[4];
#pragma unroll
        for (int j = 0; j < 4; ++j)
#pragma unroll
            for (int e = 0; e < 8; ++e) acc[j][e] = 0.0f;

#pragma unroll
        for (int c = 0; c < 4; ++c) {
#pragma unroll
            for (int j = 0; j < 4; ++j) {
                // b = K[key = j*16+l16][d = c*32 + half*8 + {0..7} and +16] from padded LDS rows
                const unsigned short* kp = kstage + (j * 16 + l16) * KSTRIDE + c * 32 + half * 8;
                Frag16 bfr;
                bfr.h[0] = *(const u16x8*)(kp);
                bfr.h[1] = *(const u16x8*)(kp + 16);
                acc[j] = wmma_bf16(qa[c].v, bfr.v, acc[j]);
            }
        }
#pragma unroll
        for (int j = 0; j < 4; ++j)
#pragma unroll
            for (int r = 0; r < 8; ++r)
                sc[(wave * 16 + half * 8 + r) * SC_STRIDE + t0 + j * 16 + l16] =
                    acc[j][r] * scale;
        __syncthreads();   // kstage reused next iteration
    }

    // ---- one-pass causal softmax (thread r owns row r) ----
    if (tid < 64) {
        const int s = (r0 + tid) >> 2;
        const int valid = s + 1;
        float* row = sc + tid * SC_STRIDE;
        float m = -3.4e38f;
        for (int t = 0; t < valid; ++t) m = fmaxf(m, row[t]);
        float sum = 0.0f;
        for (int t = 0; t < valid; ++t) { float e = __expf(row[t] - m); row[t] = e; sum += e; }
        const float inv = 1.0f / sum;
        for (int t = 0; t < valid; ++t) row[t] *= inv;
        for (int t = valid; t < nk64; ++t) row[t] = 0.0f;   // zero-mask: P@V needs no masks
    }
    __syncthreads();

    // ---- out = P @ V ----
    v8f oacc[8];
#pragma unroll
    for (int jd = 0; jd < 8; ++jd)
#pragma unroll
        for (int e = 0; e < 8; ++e) oacc[jd][e] = 0.0f;

    for (int t0 = 0; t0 < nk64; t0 += 64) {
        // stage V tile transposed: vt[d][t-t0]
        for (int c = tid; c < 64 * 16; c += 128) {
            const int t  = c >> 4;
            const int d0 = (c & 15) * 8;
            const unsigned short* vp =
                Vb + ((size_t)(b * SS + t0 + t) * HKVN + h) * DD + d0;
            u16x8 vv = *(const u16x8*)vp;
#pragma unroll
            for (int e2 = 0; e2 < 8; ++e2) vt[(d0 + e2) * VT_STRIDE + t] = vv[e2];
        }
        __syncthreads();

#pragma unroll
        for (int scn = 0; scn < 2; ++scn) {            // two 32-key chunks
            const int kc = t0 + scn * 32;
            Frag16 pa;                                  // P fragment (f32 LDS -> bf16)
            {
                const float* pr = sc + (wave * 16 + l16) * SC_STRIDE + kc + half * 8;
                float4 p0 = *(const float4*)(pr);
                float4 p1 = *(const float4*)(pr + 4);
                float4 p2 = *(const float4*)(pr + 16);
                float4 p3 = *(const float4*)(pr + 20);
                pa.e[0]  = f2bf(p0.x); pa.e[1]  = f2bf(p0.y); pa.e[2]  = f2bf(p0.z); pa.e[3]  = f2bf(p0.w);
                pa.e[4]  = f2bf(p1.x); pa.e[5]  = f2bf(p1.y); pa.e[6]  = f2bf(p1.z); pa.e[7]  = f2bf(p1.w);
                pa.e[8]  = f2bf(p2.x); pa.e[9]  = f2bf(p2.y); pa.e[10] = f2bf(p2.z); pa.e[11] = f2bf(p2.w);
                pa.e[12] = f2bf(p3.x); pa.e[13] = f2bf(p3.y); pa.e[14] = f2bf(p3.z); pa.e[15] = f2bf(p3.w);
            }
#pragma unroll
            for (int jd = 0; jd < 8; ++jd) {
                const unsigned short* vp = vt + (jd * 16 + l16) * VT_STRIDE + scn * 32 + half * 8;
                Frag16 vb_;
                vb_.h[0] = *(const u16x8*)(vp);
                vb_.h[1] = *(const u16x8*)(vp + 16);
                oacc[jd] = wmma_bf16(pa.v, vb_.v, oacc[jd]);
            }
        }
        __syncthreads();
    }

    // ---- store bf16 output rows: (s,g) -> token, hidden (h*G+g)*D + d ----
    const int rbase = r0 + wave * 16 + half * 8;
#pragma unroll
    for (int jd = 0; jd < 8; ++jd) {
        const int d = jd * 16 + l16;
#pragma unroll
        for (int r = 0; r < 8; ++r) {
            const int rr = rbase + r;
            const int s = rr >> 2, g = rr & 3;
            O[(size_t)(b * SS + s) * HIDN + (h * GG + g) * DD + d] = f2bf(oacc[jd][r]);
        }
    }
}

// ---------------- host: orchestrate on stream ----------------
extern "C" void kernel_launch(void* const* d_in, const int* in_sizes, int n_in,
                              void* d_out, int out_size, void* d_ws, size_t ws_size,
                              hipStream_t stream)
{
    const int*   positions = (const int*)d_in[0];
    const float* hidden    = (const float*)d_in[1];
    const float* w_qkv     = (const float*)d_in[2];
    const float* w_dense   = (const float*)d_in[3];
    float*       out       = (float*)d_out;

    char* ws = (char*)d_ws;
    size_t off = 0;
    auto alloc = [&](size_t bytes) -> void* {
        void* p = ws + off;
        off = (off + bytes + 255) & ~(size_t)255;
        return p;
    };

    unsigned short* hs_bf     = (unsigned short*)alloc((size_t)TT * HIDN * 2);     // 32 MB
    unsigned short* wqkv_bf   = (unsigned short*)alloc((size_t)NQKV * HIDN * 2);   // 48 MB
    unsigned short* wdense_bf = (unsigned short*)alloc((size_t)HIDN * HIDN * 2);   // 32 MB
    float*          qkv_f32   = (float*)alloc((size_t)TT * NQKV * 4);              // 96 MB
    unsigned short* q_bf      = (unsigned short*)alloc((size_t)TT * HH * DD * 2);  // 32 MB
    unsigned short* k_bf      = (unsigned short*)alloc((size_t)TT * HKVN * DD * 2);//  8 MB
    unsigned short* v_bf      = (unsigned short*)alloc((size_t)TT * HKVN * DD * 2);//  8 MB
    unsigned short* attn_bf   = (unsigned short*)qkv_f32;   // alias: qkv dead after rope

    // 1) convert inputs to bf16
    {
        int n1 = TT * HIDN;   cvt_f32_bf16_kernel<<<n1 / 1024, 256, 0, stream>>>(hidden,  hs_bf,     n1);
        int n2 = NQKV * HIDN; cvt_f32_bf16_kernel<<<n2 / 1024, 256, 0, stream>>>(w_qkv,   wqkv_bf,   n2);
        int n3 = HIDN * HIDN; cvt_f32_bf16_kernel<<<n3 / 1024, 256, 0, stream>>>(w_dense, wdense_bf, n3);
    }

    // 2) qkv = hs @ w_qkv^T   (M=4096, N=6144, K=4096)
    gemm_bf16_nt_kernel<<<dim3(NQKV / 128, TT / 128), 128, 0, stream>>>(
        hs_bf, wqkv_bf, qkv_f32, TT, NQKV, HIDN);

    // 3) RoPE + split to bf16
    rope_split_kernel<<<TT, 256, 0, stream>>>(qkv_f32, positions, q_bf, k_bf, v_bf);

    // 4) causal GQA attention (~292 KB dynamic LDS: fits CDNA5's 320 KB/WGP)
    hipFuncSetAttribute((const void*)falcon_attention_kernel,
                        hipFuncAttributeMaxDynamicSharedMemorySize, ATTN_LDS_BYTES);
    falcon_attention_kernel<<<dim3(SS * GG / 64, HKVN, BB), 128, ATTN_LDS_BYTES, stream>>>(
        q_bf, k_bf, v_bf, attn_bf);

    // 5) out = attn @ w_dense^T (M=4096, N=4096, K=4096), f32 output
    gemm_bf16_nt_kernel<<<dim3(HIDN / 128, TT / 128), 128, 0, stream>>>(
        attn_bf, wdense_bf, out, TT, HIDN, HIDN);
}
